// SpatioTemporalModel_43946105373001
// MI455X (gfx1250) — compile-verified, matrix operand order checked by reading, not run on hardware
//
#include <hip/hip_runtime.h>
#include <hip/hip_bf16.h>
#include <math.h>

#define B_ 16
#define T_ 32
#define Z_ 1024
#define F_ 32
#define HG_ 64
#define HL_ 128
#define E_ 32768
#define G_ (B_*T_)        // 512 graphs
#define N_ (B_*Z_)        // 16384 sequences
#define MROWS (G_*Z_)     // 524288 node rows

typedef __bf16 bf16;
typedef bf16 bf16x8  __attribute__((ext_vector_type(8)));
typedef bf16 bf16x16 __attribute__((ext_vector_type(16)));
typedef float f32x8  __attribute__((ext_vector_type(8)));
typedef unsigned int u32x4 __attribute__((ext_vector_type(4)));
typedef int i32x4 __attribute__((ext_vector_type(4)));
typedef int i32x8 __attribute__((ext_vector_type(8)));

// ---------------------------------------------------------------------------
// WMMA bf16 16x16x32 fragment load from LDS.
// A-matrix layout (ISA 7.12.2, 16-bit A 16x32): lane<16 holds row (lane),
// K = {0..7, 16..23}; lane>=16 holds row (lane-16), K = {8..15, 24..31}.
// B uses the symmetric layout over columns, so we stage B transposed
// (column-major: [n][k]) and reuse the same loader.
// ---------------------------------------------------------------------------
static __device__ __forceinline__ bf16x16 frag_ld(const bf16* lds, int rc, int ldk, int koff) {
  const int lane = threadIdx.x & 31;
  const int i = lane & 15;
  const int kh = (lane & 16) ? 8 : 0;
  const bf16* p = lds + (rc + i) * ldk + koff + kh;
  bf16x8 lo = *(const bf16x8*)(p);
  bf16x8 hi = *(const bf16x8*)(p + 16);
  return __builtin_shufflevector(lo, hi, 0,1,2,3,4,5,6,7,8,9,10,11,12,13,14,15);
}

static __device__ __forceinline__ float sigm(float x) { return 1.f / (1.f + expf(-x)); }

// ---------------------------------------------------------------------------
// Tensor Data Mover: async 2D tile load (rows x tile_k bf16, row stride ktot
// elements) from global into LDS. Descriptor per ISA ch.8 (type=2 "image",
// data_size=1 -> 2 bytes). Issued by one wave; completion via TENSORcnt.
// ---------------------------------------------------------------------------
static __device__ __forceinline__ void tdm_load_2d_bf16(
    unsigned lds_off, const void* gptr, unsigned ktot, unsigned rows, unsigned tile_k) {
  unsigned long long ga = (unsigned long long)gptr;
  u32x4 g0;
  g0[0] = 1u;                                            // count=1, user D#
  g0[1] = lds_off;                                       // lds_addr (bytes)
  g0[2] = (unsigned)(ga & 0xFFFFFFFFu);                  // global_addr[31:0]
  g0[3] = (unsigned)((ga >> 32) & 0x1FFFFFFu) | (2u << 30); // addr[56:32] | type=2
  i32x8 g1;
  g1[0] = (int)(1u << 16);                               // data_size=1 (2B), mask=0
  g1[1] = (int)((ktot & 0xFFFFu) << 16);                 // tensor_dim0[15:0]
  g1[2] = (int)(((ktot >> 16) & 0xFFFFu) | ((rows & 0xFFFFu) << 16)); // dim0 hi | dim1 lo
  g1[3] = (int)(((rows >> 16) & 0xFFFFu) | ((tile_k & 0xFFFFu) << 16)); // dim1 hi | tile_dim0
  g1[4] = (int)(rows & 0xFFFFu);                         // tile_dim1 | tile_dim2=0
  g1[5] = (int)ktot;                                     // tensor_dim0_stride[31:0]
  g1[6] = 0;                                             // stride hi | dim1_stride lo
  g1[7] = 0;
  i32x4 z4 = {0, 0, 0, 0};
#if defined(__clang_major__) && (__clang_major__ >= 23)
  i32x8 z8 = {0, 0, 0, 0, 0, 0, 0, 0};
  __builtin_amdgcn_tensor_load_to_lds(g0, g1, z4, z4, z8, 0);
#else
  __builtin_amdgcn_tensor_load_to_lds(g0, g1, z4, z4, 0);
#endif
}

// ---------------------------------------------------------------------------
// CSR build (edge list shared by all 512 graphs -> build once)
// ---------------------------------------------------------------------------
__global__ void csr_count(const int* __restrict__ ei, const float* __restrict__ ew,
                          int* __restrict__ cnt, float* __restrict__ degw) {
  int e = blockIdx.x * blockDim.x + threadIdx.x;
  if (e < E_) {
    int d = ei[E_ + e];               // dst row of edge_index [2, E]
    atomicAdd(&cnt[d], 1);
    atomicAdd(&degw[d], ew[e]);       // weighted in-degree (shared across graphs)
  }
}

__global__ __launch_bounds__(1024) void csr_scan(const int* __restrict__ cnt,
                                                 int* __restrict__ row_start) {
  __shared__ int buf[Z_];
  int t = threadIdx.x;
  buf[t] = cnt[t];
  __syncthreads();
  for (int off = 1; off < Z_; off <<= 1) {
    int v = (t >= off) ? buf[t - off] : 0;
    __syncthreads();
    buf[t] += v;
    __syncthreads();
  }
  row_start[t + 1] = buf[t];
  if (t == 0) row_start[0] = 0;
}

__global__ void csr_fill(const int* __restrict__ ei, const float* __restrict__ ew,
                         const int* __restrict__ row_start, int* __restrict__ cursor,
                         int* __restrict__ csr_src, float* __restrict__ csr_w) {
  int e = blockIdx.x * blockDim.x + threadIdx.x;
  if (e < E_) {
    int d = ei[E_ + e];
    int p = atomicAdd(&cursor[d], 1);
    int idx = row_start[d] + p;
    csr_src[idx] = ei[e];             // src row
    csr_w[idx] = ew[e];
  }
}

// ---------------------------------------------------------------------------
// Pack LSTM weights once: WT[n][k] (bf16, gate-column major, k = [Wih | Whh])
// so the step kernel's B tile is a contiguous-row 2D tile fit for the TDM.
// ---------------------------------------------------------------------------
__global__ void pack_weights(const float* __restrict__ Wih, const float* __restrict__ Whh,
                             bf16* __restrict__ WT, int Kx) {
  int Ktot = Kx + HL_;
  int total = 4 * HL_ * Ktot;   // 512 * Ktot
  for (int i = blockIdx.x * blockDim.x + threadIdx.x; i < total; i += gridDim.x * blockDim.x) {
    int n = i / Ktot, k = i - n * Ktot;
    float v = (k < Kx) ? Wih[n * Kx + k] : Whh[n * HL_ + (k - Kx)];
    WT[i] = (bf16)v;
  }
}

// ---------------------------------------------------------------------------
// Dense GEMM + bias: out[rows,64] = in[rows,K] @ W[K,64] + b   (K = 32 or 64)
// Block: 256 threads = 8 waves; tile 128 rows x 64 cols; wave = 16 rows x 64.
// ---------------------------------------------------------------------------
__global__ __launch_bounds__(256) void gemm_bias(
    const float* __restrict__ in, const float* __restrict__ W,
    const float* __restrict__ bias, float* __restrict__ out, int K) {
  extern __shared__ char smem[];
  bf16* ldsA  = (bf16*)smem;                         // 128 x K
  bf16* ldsBT = (bf16*)(smem + (size_t)128 * K * 2); // 64 x K (col-major weights)
  const int tid = threadIdx.x;
  const long long row0 = (long long)blockIdx.x * 128;
  const int shift = (K == 32) ? 5 : 6;

  const float* inb = in + row0 * K;
  for (int i = tid; i < 128 * K; i += 256) ldsA[i] = (bf16)inb[i];
  for (int i = tid; i < 64 * K; i += 256) {
    int n = i >> shift, k = i & (K - 1);
    ldsBT[n * K + k] = (bf16)W[k * 64 + n];
  }
  __syncthreads();

  const int w = tid >> 5, lane = tid & 31;
  const int rb = w * 16;
  f32x8 acc[4] = {};
  for (int kc = 0; kc < K; kc += 32) {
    bf16x16 a = frag_ld(ldsA, rb, K, kc);
#pragma unroll
    for (int nt = 0; nt < 4; ++nt) {
      bf16x16 b = frag_ld(ldsBT, nt * 16, K, kc);
      acc[nt] = __builtin_amdgcn_wmma_f32_16x16x32_bf16(false, a, false, b,
                                                        (short)0, acc[nt], false, false);
    }
  }
  const int mh = (lane & 16) ? 8 : 0;
  const int nl = lane & 15;
#pragma unroll
  for (int nt = 0; nt < 4; ++nt)
#pragma unroll
    for (int v = 0; v < 8; ++v) {
      long long r = row0 + rb + v + mh;
      int n = nt * 16 + nl;
      out[r * 64 + n] = acc[nt][v] + bias[n];
    }
}

// ---------------------------------------------------------------------------
// GCN aggregate: out = relu(h + (CSR-gather of ew*h[src]) / degw), optional
// fused LayerNorm over the 64 features. One block of 64 threads per (g, z).
// ---------------------------------------------------------------------------
__global__ __launch_bounds__(64) void gcn_aggregate(
    const float* __restrict__ h, float* __restrict__ out,
    const int* __restrict__ row_start, const int* __restrict__ csr_src,
    const float* __restrict__ csr_w, const float* __restrict__ degw,
    const float* __restrict__ ln_g, const float* __restrict__ ln_b, int do_ln) {
  __shared__ float red[64];
  const int gid = blockIdx.x;
  const int g = gid >> 10, z = gid & (Z_ - 1);
  const int f = threadIdx.x;
  const float* hg = h + (long long)g * Z_ * HG_;

  const int e0 = row_start[z], e1 = row_start[z + 1];
  float acc = 0.f;
  for (int e = e0; e < e1; ++e) {
    int s = csr_src[e];
    if (e + 1 < e1) __builtin_prefetch(&hg[csr_src[e + 1] * HG_ + f], 0, 1);
    acc = fmaf(csr_w[e], hg[s * HG_ + f], acc);
  }
  float d = fmaxf(degw[z], 1e-6f);
  float v = fmaxf(hg[z * HG_ + f] + acc / d, 0.f);   // relu

  if (do_ln) {
    red[f] = v; __syncthreads();
    for (int s2 = 32; s2 > 0; s2 >>= 1) { if (f < s2) red[f] += red[f + s2]; __syncthreads(); }
    float mu = red[0] * (1.f / HG_);
    __syncthreads();
    float dv = v - mu;
    red[f] = dv * dv; __syncthreads();
    for (int s2 = 32; s2 > 0; s2 >>= 1) { if (f < s2) red[f] += red[f + s2]; __syncthreads(); }
    float var = red[0] * (1.f / HG_);
    v = dv * rsqrtf(var + 1e-5f) * ln_g[f] + ln_b[f];
  }
  out[(long long)g * Z_ * HG_ + z * HG_ + f] = v;
}

// ---------------------------------------------------------------------------
// One LSTM timestep for one layer, fully fused:
//   gates[32 rows x 512] = x_t @ Wih^T + h @ Whh^T + (bih + bhh)  (WMMA bf16)
//   then pointwise cell update through LDS. Weights arrive pre-packed bf16
//   (WT[n][Ktot]); each K-chunk's 512x32 B tile is DMA'd into LDS by the TDM
//   (wave 0 issues, TENSORcnt waits), overlapped with cooperative A staging.
// seq_mode 0: x_t comes from the GCN output [b*T+t][z][HG] (n = b*Z + z).
// seq_mode 1: x_t = contiguous [N, Kx] (previous layer's h).
// ---------------------------------------------------------------------------
__global__ __launch_bounds__(256) void lstm_step(
    const float* __restrict__ xin, int seq_mode, int t,
    const bf16* __restrict__ WT,
    const float* __restrict__ bih, const float* __restrict__ bhh,
    float* __restrict__ h, float* __restrict__ c, int Kx) {
  extern __shared__ char smem[];
  bf16* ldsA   = (bf16*)smem;                      // 32 x 32
  bf16* ldsBT  = (bf16*)(smem + 2048);             // 512 x 32 (gate-col major)
  float* ldsG  = (float*)(smem + 2048 + 32768);    // 32 x 512 gates

  const int tid = threadIdx.x;
  const int row0 = blockIdx.x * 32;
  const int w = tid >> 5, lane = tid & 31;
  const int rg = (w & 1) * 16;       // row group within tile
  const int cg = (w >> 1) * 128;     // 128 gate cols per wave
  const unsigned ldsBT_off = (unsigned)(unsigned long long)ldsBT; // flat low 32 = LDS offset
  f32x8 acc[8] = {};

  const int Ktot = Kx + HL_;
  for (int cb = 0; cb < Ktot; cb += 32) {
    // wave 0 kicks the async B-tile DMA: 512 rows x 32 bf16, row stride Ktot
    if (w == 0) tdm_load_2d_bf16(ldsBT_off, (const void*)(WT + cb),
                                 (unsigned)Ktot, 512u, 32u);
    // all waves: stage A (activations) chunk, f32 -> bf16
    for (int i = tid; i < 32 * 32; i += 256) {
      int r = i >> 5, kk = i & 31;
      int n = row0 + r;
      int kg = cb + kk;
      float vv;
      if (kg < Kx) {
        if (seq_mode == 0) {
          int b = n >> 10, z = n & (Z_ - 1);
          vv = xin[(((long long)b * T_ + t) * Z_ + z) * HG_ + kg];
        } else {
          vv = xin[(long long)n * Kx + kg];
        }
      } else {
        vv = h[(long long)n * HL_ + (kg - Kx)];
      }
      ldsA[r * 32 + kk] = (bf16)vv;
    }
    if (w == 0) __builtin_amdgcn_s_wait_tensorcnt((short)0);
    __syncthreads();

    bf16x16 a = frag_ld(ldsA, rg, 32, 0);
#pragma unroll
    for (int ct = 0; ct < 8; ++ct) {
      bf16x16 b = frag_ld(ldsBT, cg + ct * 16, 32, 0);
      acc[ct] = __builtin_amdgcn_wmma_f32_16x16x32_bf16(false, a, false, b,
                                                        (short)0, acc[ct], false, false);
    }
    __syncthreads();   // everyone done reading ldsA/ldsBT before next chunk
  }

  // spill gates (+ combined bias) to LDS so i/f/g/o can be re-grouped
  const int mh = (lane & 16) ? 8 : 0;
  const int nl = lane & 15;
#pragma unroll
  for (int ct = 0; ct < 8; ++ct)
#pragma unroll
    for (int v = 0; v < 8; ++v) {
      int r = rg + v + mh;
      int col = cg + ct * 16 + nl;
      ldsG[r * 512 + col] = acc[ct][v] + bih[col] + bhh[col];
    }
  __syncthreads();

  // pointwise LSTM cell (PyTorch gate order i, f, g, o)
  for (int i = tid; i < 32 * HL_; i += 256) {
    int r = i >> 7, j = i & (HL_ - 1);
    long long n = row0 + r;
    float gi = ldsG[r * 512 + j];
    float gf = ldsG[r * 512 + 128 + j];
    float gg = ldsG[r * 512 + 256 + j];
    float go = ldsG[r * 512 + 384 + j];
    float cv = sigm(gf) * c[n * HL_ + j] + sigm(gi) * tanhf(gg);
    c[n * HL_ + j] = cv;
    h[n * HL_ + j] = sigm(go) * tanhf(cv);
  }
}

// ---------------------------------------------------------------------------
// Head: out[n] = relu(h1[n] @ Wh1 + bh1) @ Wh2 + bh2   (tiny, f32 VALU)
// ---------------------------------------------------------------------------
__global__ __launch_bounds__(64) void head_kernel(
    const float* __restrict__ h1, const float* __restrict__ Wh1,
    const float* __restrict__ bh1, const float* __restrict__ Wh2,
    const float* __restrict__ bh2, float* __restrict__ out) {
  __shared__ float red[64];
  int n = blockIdx.x, j = threadIdx.x;
  const float* hr = h1 + (long long)n * HL_;
  float a = bh1[j];
  for (int k = 0; k < HL_; ++k) a = fmaf(hr[k], Wh1[k * 64 + j], a);
  a = fmaxf(a, 0.f);
  red[j] = a * Wh2[j];
  __syncthreads();
  for (int s = 32; s > 0; s >>= 1) { if (j < s) red[j] += red[j + s]; __syncthreads(); }
  if (j == 0) out[n] = red[0] + bh2[0];
}

// ---------------------------------------------------------------------------
extern "C" void kernel_launch(void* const* d_in, const int* in_sizes, int n_in,
                              void* d_out, int out_size, void* d_ws, size_t ws_size,
                              hipStream_t stream) {
  (void)in_sizes; (void)n_in; (void)out_size; (void)ws_size;
  const float* x    = (const float*)d_in[0];
  const int*   ei   = (const int*)d_in[1];
  const float* ew   = (const float*)d_in[2];
  const float* W1   = (const float*)d_in[3];
  const float* b1   = (const float*)d_in[4];
  const float* W2   = (const float*)d_in[5];
  const float* b2   = (const float*)d_in[6];
  const float* lng  = (const float*)d_in[7];
  const float* lnb  = (const float*)d_in[8];
  const float* Wih0 = (const float*)d_in[9];
  const float* Whh0 = (const float*)d_in[10];
  const float* bih0 = (const float*)d_in[11];
  const float* bhh0 = (const float*)d_in[12];
  const float* Wih1 = (const float*)d_in[13];
  const float* Whh1 = (const float*)d_in[14];
  const float* bih1 = (const float*)d_in[15];
  const float* bhh1 = (const float*)d_in[16];
  const float* Wh1  = (const float*)d_in[17];
  const float* bh1  = (const float*)d_in[18];
  const float* Wh2  = (const float*)d_in[19];
  const float* bh2  = (const float*)d_in[20];
  float* outp = (float*)d_out;

  char* ws = (char*)d_ws;
  const size_t HBYTES = (size_t)MROWS * HG_ * 4;     // 128 MB per buffer
  float* hA = (float*)ws;
  float* hB = (float*)(ws + HBYTES);
  // LSTM state reuses the hA region (free after the last GEMM+aggregate)
  const size_t ST = (size_t)N_ * HL_ * 4;            // 8 MB per state tensor
  float* h0 = (float*)ws;
  float* c0 = (float*)(ws + ST);
  float* h1 = (float*)(ws + 2 * ST);
  float* c1 = (float*)(ws + 3 * ST);
  // small region after the two big buffers: CSR + packed bf16 LSTM weights
  char* S = ws + 2 * HBYTES;
  int*   cnt    = (int*)(S);
  int*   cursor = (int*)(S + 4096);
  float* degw   = (float*)(S + 8192);
  int*   rowst  = (int*)(S + 12288);
  int*   csrsrc = (int*)(S + 20480);
  float* csrw   = (float*)(S + 20480 + (size_t)E_ * 4);
  bf16*  WT0    = (bf16*)(S + 282624);               // 512 x 192 bf16 (192 KB)
  bf16*  WT1    = (bf16*)(S + 479232);               // 512 x 256 bf16 (256 KB)

  // --- build CSR by destination (once; shared by all 512 graphs) ---
  hipMemsetAsync(S, 0, 12288, stream);
  csr_count<<<E_ / 256, 256, 0, stream>>>(ei, ew, cnt, degw);
  csr_scan<<<1, 1024, 0, stream>>>(cnt, rowst);
  csr_fill<<<E_ / 256, 256, 0, stream>>>(ei, ew, rowst, cursor, csrsrc, csrw);

  // --- pack LSTM weights to bf16 gate-column-major (for TDM B tiles) ---
  pack_weights<<<96, 256, 0, stream>>>(Wih0, Whh0, WT0, HG_);
  pack_weights<<<128, 256, 0, stream>>>(Wih1, Whh1, WT1, HL_);

  // --- GCN layer 1: GEMM -> aggregate ---
  gemm_bias<<<MROWS / 128, 256, 384 * 32, stream>>>(x, W1, b1, hA, 32);
  gcn_aggregate<<<MROWS, 64, 0, stream>>>(hA, hB, rowst, csrsrc, csrw, degw, lng, lnb, 0);
  // --- GCN layer 2: GEMM -> aggregate (+fused LayerNorm) ---
  gemm_bias<<<MROWS / 128, 256, 384 * 64, stream>>>(hB, W2, b2, hA, 64);
  gcn_aggregate<<<MROWS, 64, 0, stream>>>(hA, hB, rowst, csrsrc, csrw, degw, lng, lnb, 1);

  // --- 2-layer LSTM, pipelined per timestep ---
  hipMemsetAsync(ws, 0, 4 * ST, stream);             // zero h0,c0,h1,c1
  const int LSTM_LDS = 2048 + 32768 + 65536;         // A + B^T + gates
  for (int t = 0; t < T_; ++t) {
    lstm_step<<<N_ / 32, 256, LSTM_LDS, stream>>>(hB, 0, t, WT0, bih0, bhh0, h0, c0, HG_);
    lstm_step<<<N_ / 32, 256, LSTM_LDS, stream>>>(h0, 1, t, WT1, bih1, bhh1, h1, c1, HL_);
  }

  // --- head MLP ---
  head_kernel<<<N_, 64, 0, stream>>>(h1, Wh1, bh1, Wh2, bh2, outp);
}
